// CrossAttentionBlock_80985903333480
// MI455X (gfx1250) — matrix operand order repaired and checked by reading, not compiled, not run on hardware
//
#include <hip/hip_runtime.h>

// ---------------------------------------------------------------------------
// Types for CDNA5 WMMA (wave32): bf16 16x16x32 -> f32 accumulate
// ---------------------------------------------------------------------------
typedef __bf16 bf16;
typedef __attribute__((ext_vector_type(16))) __bf16 bf16x16;
typedef __attribute__((ext_vector_type(8)))  float   f32x8;

union FragB { bf16x16 v; uint4 u[2]; };

__device__ inline f32x8 fzero8() {
  f32x8 z;
#pragma unroll
  for (int i = 0; i < 8; ++i) z[i] = 0.0f;
  return z;
}

__device__ inline float gelu_exact(float x) {
  return 0.5f * x * (1.0f + erff(x * 0.70710678118654752f));
}

// CDNA5 async global->LDS copy (ASYNCcnt-tracked, bypasses VGPRs).
// vdst = 32-bit LDS byte offset, addr = 64-bit global address (VGPR pair).
// NOTE (ISA 15.18): INST_OFFSET is added to BOTH the LDS and global address,
// so one (ldsoff, gptr) pair can service several 16B chunks of a row.
#define ASYNC_B128(ldsoff, gptr, imm)                                  \
  asm volatile("global_load_async_to_lds_b128 %0, %1, off offset:%2"   \
               :: "v"(ldsoff), "v"(gptr), "i"(imm) : "memory")
#define WAIT_ASYNC() asm volatile("s_wait_asynccnt 0x0" ::: "memory")

// ---------------------------------------------------------------------------
// Constants of the problem
// ---------------------------------------------------------------------------
#define BB   4
#define TQ_  4096
#define TB_  2048
#define CC   512
#define HH   8
#define DD   64
#define INNER_ 2048
#define KCH  8          // K-split chunks for ctx kernel

// ---------------------------------------------------------------------------
// Weight convert + transpose: w[din][dout] f32 -> wT[dout][din] bf16
// ---------------------------------------------------------------------------
__global__ __launch_bounds__(256)
void wt_kernel(const float* __restrict__ w, bf16* __restrict__ wT,
               int din, int dout) {
  int i = blockIdx.x * 256 + threadIdx.x;
  if (i >= din * dout) return;
  int r = i / dout, c = i % dout;
  wT[(size_t)c * din + r] = (bf16)w[i];
}

// f32 -> bf16 elementwise
__global__ __launch_bounds__(256)
void cvt_bf16_kernel(const float* __restrict__ in, bf16* __restrict__ out, long n) {
  long i = (long)blockIdx.x * 256 + threadIdx.x;
  if (i < n) out[i] = (bf16)in[i];
}

__global__ __launch_bounds__(256)
void zero_kernel(float* __restrict__ p, int n) {
  int i = blockIdx.x * 256 + threadIdx.x;
  if (i < n) p[i] = 0.0f;
}

// ---------------------------------------------------------------------------
// LayerNorm over C=512, one wave (32 lanes) per row, f32 in -> bf16 out
// ---------------------------------------------------------------------------
__global__ __launch_bounds__(256)
void ln_kernel(const float* __restrict__ x, const float* __restrict__ g,
               const float* __restrict__ bta, bf16* __restrict__ out) {
  const int row  = blockIdx.x * 8 + (threadIdx.x >> 5);
  const int lane = threadIdx.x & 31;
  const float* xr = x + (size_t)row * CC;
  float vals[16];
  float s = 0.f, s2 = 0.f;
#pragma unroll
  for (int i = 0; i < 16; ++i) {
    float v = xr[lane + 32 * i];
    vals[i] = v; s += v; s2 += v * v;
  }
#pragma unroll
  for (int m = 16; m > 0; m >>= 1) {
    s  += __shfl_xor(s,  m, 32);
    s2 += __shfl_xor(s2, m, 32);
  }
  float mean = s * (1.0f / CC);
  float var  = s2 * (1.0f / CC) - mean * mean;
  float rstd = rsqrtf(var + 1e-5f);
  bf16* orow = out + (size_t)row * CC;
#pragma unroll
  for (int i = 0; i < 16; ++i) {
    int c = lane + 32 * i;
    orow[c] = (bf16)((vals[i] - mean) * rstd * g[c] + bta[c]);
  }
}

// ---------------------------------------------------------------------------
// Softmax over contiguous 64-element head groups; one wave per group.
// ---------------------------------------------------------------------------
__global__ __launch_bounds__(256)
void softmax64_kernel(const float* __restrict__ in, bf16* __restrict__ outb,
                      float* __restrict__ outf) {
  const size_t grp  = (size_t)blockIdx.x * 8 + (threadIdx.x >> 5);
  const int    lane = threadIdx.x & 31;
  const size_t base = grp * 64;
  float v0 = in[base + lane], v1 = in[base + 32 + lane];
  float mx = fmaxf(v0, v1);
#pragma unroll
  for (int m = 16; m > 0; m >>= 1) mx = fmaxf(mx, __shfl_xor(mx, m, 32));
  float e0 = __expf(v0 - mx), e1 = __expf(v1 - mx);
  float s = e0 + e1;
#pragma unroll
  for (int m = 16; m > 0; m >>= 1) s += __shfl_xor(s, m, 32);
  float r = 1.0f / s;
  e0 *= r; e1 *= r;
  outb[base + lane]      = (bf16)e0;
  outb[base + 32 + lane] = (bf16)e1;
  if (outf) { outf[base + lane] = e0; outf[base + 32 + lane] = e1; }
}

// ---------------------------------------------------------------------------
// Column sums of k over T rows per batch (T-chunked + atomics)
// ---------------------------------------------------------------------------
__global__ __launch_bounds__(256)
void ksum_kernel(const bf16* __restrict__ k, float* __restrict__ ks, int T) {
  int i = blockIdx.x * 256 + threadIdx.x;          // i over BB*CC
  int b = i / CC, c = i % CC;
  int tchunk = T / gridDim.y;
  int t0 = blockIdx.y * tchunk;
  float s = 0.f;
  for (int t = t0; t < t0 + tchunk; ++t)
    s += (float)k[((size_t)b * T + t) * CC + c];
  atomicAdd(&ks[(size_t)b * CC + c], s);
}

// d_inv[b,h,t] = 1/clip(sum_d q[b,t,h*64+d]*ksum[b,h*64+d], clipmin)
__global__ __launch_bounds__(256)
void dinv_kernel(const bf16* __restrict__ q, const float* __restrict__ ks,
                 float* __restrict__ dinv, float clipmin) {
  int i = blockIdx.x * 256 + threadIdx.x;          // i over BB*HH*TQ_
  int b = i / (HH * TQ_);
  int rem = i % (HH * TQ_);
  int h = rem / TQ_, t = rem % TQ_;
  const bf16* qr = q + ((size_t)b * TQ_ + t) * CC + h * DD;
  const float* kr = ks + (size_t)b * CC + h * DD;
  float s = 0.f;
#pragma unroll
  for (int d = 0; d < DD; ++d) s += (float)qr[d] * kr[d];
  dinv[i] = 1.0f / fmaxf(s, clipmin);
}

// ---------------------------------------------------------------------------
// Generic bf16 WMMA GEMM: C[M,N] = A[M,K] * BT[N,K]^T + bias (+opts)
// 256 threads = 8 waves; 128x128 block tile; 64x32 wave tile; BK=64
// (2 WMMA k-steps = 16 WMMAs per staged slice). Async double-buffered
// global->LDS staging (ASYNCcnt).
// ---------------------------------------------------------------------------
template <int OUT_BF16, int DO_GELU, int DO_RESID>
__global__ __launch_bounds__(256)
void gemm128(const bf16* __restrict__ A, const bf16* __restrict__ BT,
             const float* __restrict__ bias, const float* __restrict__ resid,
             float* __restrict__ Cf, bf16* __restrict__ Cb,
             int M, int N, int K) {
  __shared__ bf16 As[2][128][64];
  __shared__ bf16 Bs[2][128][64];
  const int tid  = threadIdx.x;
  const int wave = tid >> 5;
  const int lane = tid & 31;
  const int half = lane >> 4;
  const int lrow = lane & 15;
  const int wm0  = (wave >> 2) * 64;   // wave row base in block tile
  const int wn0  = (wave & 3) * 32;    // wave col base in block tile
  const int m0   = blockIdx.y * 128;
  const int n0   = blockIdx.x * 128;

  // staging assignment: thread -> rows (srow, srow+64), cols seg8 & seg8+32
  const int srow = tid >> 2;
  const int seg8 = (tid & 3) * 8;
  const bf16* pA0 = A  + (size_t)(m0 + srow) * K + seg8;
  const bf16* pA1 = pA0 + (size_t)64 * K;
  const bf16* pB0 = BT + (size_t)(n0 + srow) * K + seg8;
  const bf16* pB1 = pB0 + (size_t)64 * K;
  uint32_t la0[2], la1[2], lb0[2], lb1[2];
#pragma unroll
  for (int p = 0; p < 2; ++p) {
    la0[p] = (uint32_t)(uintptr_t)&As[p][srow][seg8];
    la1[p] = (uint32_t)(uintptr_t)&As[p][srow + 64][seg8];
    lb0[p] = (uint32_t)(uintptr_t)&Bs[p][srow][seg8];
    lb1[p] = (uint32_t)(uintptr_t)&Bs[p][srow + 64][seg8];
  }

  f32x8 acc[4][2];
#pragma unroll
  for (int i = 0; i < 4; ++i)
#pragma unroll
    for (int j = 0; j < 2; ++j) acc[i][j] = fzero8();

  // prologue: stage K-slice 0 into buffer 0 (offset:64 = +32 cols, both spaces)
  ASYNC_B128(la0[0], pA0, 0); ASYNC_B128(la0[0], pA0, 64);
  ASYNC_B128(la1[0], pA1, 0); ASYNC_B128(la1[0], pA1, 64);
  ASYNC_B128(lb0[0], pB0, 0); ASYNC_B128(lb0[0], pB0, 64);
  ASYNC_B128(lb1[0], pB1, 0); ASYNC_B128(lb1[0], pB1, 64);
  pA0 += 64; pA1 += 64; pB0 += 64; pB1 += 64;

  const int nsteps = K >> 6;
  for (int s = 0; s < nsteps; ++s) {
    const int p = s & 1;
    WAIT_ASYNC();            // my async copies for buffer p have landed
    __syncthreads();         // everyone's have landed; buffer p^1 free
    if (s + 1 < nsteps) {    // kick off next slice while we compute
      const int np = p ^ 1;
      ASYNC_B128(la0[np], pA0, 0); ASYNC_B128(la0[np], pA0, 64);
      ASYNC_B128(la1[np], pA1, 0); ASYNC_B128(la1[np], pA1, 64);
      ASYNC_B128(lb0[np], pB0, 0); ASYNC_B128(lb0[np], pB0, 64);
      ASYNC_B128(lb1[np], pB1, 0); ASYNC_B128(lb1[np], pB1, 64);
      pA0 += 64; pA1 += 64; pB0 += 64; pB1 += 64;
    }

#pragma unroll
    for (int kk = 0; kk < 64; kk += 32) {
      FragB bfrag[2];
#pragma unroll
      for (int tn = 0; tn < 2; ++tn) {
        int br = wn0 + tn * 16 + lrow;
        bfrag[tn].u[0] = *(const uint4*)&Bs[p][br][kk + half * 8];
        bfrag[tn].u[1] = *(const uint4*)&Bs[p][br][kk + 16 + half * 8];
      }
#pragma unroll
      for (int tm = 0; tm < 4; ++tm) {
        int ar = wm0 + tm * 16 + lrow;
        FragB afrag;
        afrag.u[0] = *(const uint4*)&As[p][ar][kk + half * 8];
        afrag.u[1] = *(const uint4*)&As[p][ar][kk + 16 + half * 8];
#pragma unroll
        for (int tn = 0; tn < 2; ++tn)
          acc[tm][tn] = __builtin_amdgcn_wmma_f32_16x16x32_bf16(
              false, afrag.v, false, bfrag[tn].v, (short)0, acc[tm][tn],
              false, false);
      }
    }
  }

#pragma unroll
  for (int tm = 0; tm < 4; ++tm)
#pragma unroll
    for (int tn = 0; tn < 2; ++tn)
#pragma unroll
      for (int r = 0; r < 8; ++r) {
        int row = m0 + wm0 + tm * 16 + half * 8 + r;
        int col = n0 + wn0 + tn * 16 + lrow;
        float v = acc[tm][tn][r] + bias[col];
        if (DO_GELU)  v = gelu_exact(v);
        if (DO_RESID) v += resid[(size_t)row * N + col];
        size_t idx = (size_t)row * N + col;
        if (OUT_BF16) Cb[idx] = (bf16)v; else Cf[idx] = v;
      }
}

// ---------------------------------------------------------------------------
// ctx kernel: per (b,h): ctx = k^T @ v, M=N=64, K=T.  K split over
// blockIdx.y (KCH chunks -> BB*HH*KCH blocks); partial d x e tiles are
// atomically accumulated into f32 ctx^T [bh][e][d].  128 threads = 4 waves.
// ---------------------------------------------------------------------------
__global__ __launch_bounds__(128)
void ctx_kernel(const bf16* __restrict__ k, const bf16* __restrict__ v,
                float* __restrict__ ctxf, int T) {
  __shared__ bf16 Ak[64][32];
  __shared__ bf16 Bv[64][32];
  const int bh = blockIdx.x;
  const int b = bh / HH, h = bh % HH;
  const int tid = threadIdx.x, wave = tid >> 5, lane = tid & 31;
  const int half = lane >> 4, lrow = lane & 15;
  const int tchunk = T / KCH;
  const int tbeg = blockIdx.y * tchunk;

  // hoisted staging addresses: fixed column c, marching row pointer
  const int c   = tid & 63;
  const int tl0 = tid >> 6;                  // 0 or 1, rows step 2
  const bf16* kp = k + ((size_t)b * T + tbeg + tl0) * CC + h * DD + c;
  const bf16* vp = v + ((size_t)b * T + tbeg + tl0) * CC + h * DD + c;
  bf16* ak = &Ak[c][tl0];
  bf16* bv = &Bv[c][tl0];

  f32x8 acc[4];
#pragma unroll
  for (int i = 0; i < 4; ++i) acc[i] = fzero8();

  for (int t0 = 0; t0 < tchunk; t0 += 32) {
    __syncthreads();
#pragma unroll
    for (int j = 0; j < 16; ++j) {         // constant offsets after hoisting
      ak[2 * j] = kp[(size_t)(2 * j) * CC];
      bv[2 * j] = vp[(size_t)(2 * j) * CC];
    }
    kp += (size_t)32 * CC;
    vp += (size_t)32 * CC;
    __syncthreads();

    FragB afrag;
    afrag.u[0] = *(const uint4*)&Ak[wave * 16 + lrow][half * 8];
    afrag.u[1] = *(const uint4*)&Ak[wave * 16 + lrow][16 + half * 8];
#pragma unroll
    for (int tn = 0; tn < 4; ++tn) {
      FragB bfrag;
      bfrag.u[0] = *(const uint4*)&Bv[tn * 16 + lrow][half * 8];
      bfrag.u[1] = *(const uint4*)&Bv[tn * 16 + lrow][16 + half * 8];
      acc[tn] = __builtin_amdgcn_wmma_f32_16x16x32_bf16(
          false, afrag.v, false, bfrag.v, (short)0, acc[tn], false, false);
    }
  }
#pragma unroll
  for (int tn = 0; tn < 4; ++tn)
#pragma unroll
    for (int r = 0; r < 8; ++r) {
      int m = wave * 16 + half * 8 + r;   // d index
      int n = tn * 16 + lrow;             // e index
      atomicAdd(&ctxf[(size_t)bh * 64 * 64 + n * 64 + m], acc[tn][r]);
    }
}

// ---------------------------------------------------------------------------
// apply kernel: out[b,t,h*64+e] += (q[b,t,h,:] @ ctx) * dinv[b,h,t]
// K=64, N=64; 8 waves each own 16 rows x 64 cols. Async one-shot staging.
// ---------------------------------------------------------------------------
__global__ __launch_bounds__(256)
void attn_apply_kernel(const bf16* __restrict__ q, const bf16* __restrict__ ctxT,
                       const float* __restrict__ dinv, float* __restrict__ outacc) {
  __shared__ bf16 As[128][64];
  __shared__ bf16 Bs[64][64];
  const int b = blockIdx.z, h = blockIdx.y;
  const int t0 = blockIdx.x * 128;
  const int tid = threadIdx.x, wave = tid >> 5, lane = tid & 31;
  const int half = lane >> 4, lrow = lane & 15;
  const bf16* Aq = q + ((size_t)b * TQ_) * CC + h * DD;
  const bf16* Bc = ctxT + ((size_t)(b * HH + h)) * 64 * 64;

  for (int i = tid; i < 128 * 8; i += 256) {
    int row = i >> 3, sg = (i & 7) * 8;
    uint32_t lo = (uint32_t)(uintptr_t)&As[row][sg];
    const bf16* gp = &Aq[(size_t)(t0 + row) * CC + sg];
    ASYNC_B128(lo, gp, 0);
  }
  for (int i = tid; i < 64 * 8; i += 256) {
    int row = i >> 3, sg = (i & 7) * 8;
    uint32_t lo = (uint32_t)(uintptr_t)&Bs[row][sg];
    const bf16* gp = &Bc[row * 64 + sg];
    ASYNC_B128(lo, gp, 0);
  }
  WAIT_ASYNC();
  __syncthreads();

  f32x8 acc[4];
#pragma unroll
  for (int i = 0; i < 4; ++i) acc[i] = fzero8();

#pragma unroll
  for (int kk = 0; kk < 64; kk += 32) {
    FragB afrag;
    afrag.u[0] = *(const uint4*)&As[wave * 16 + lrow][kk + half * 8];
    afrag.u[1] = *(const uint4*)&As[wave * 16 + lrow][kk + 16 + half * 8];
#pragma unroll
    for (int tn = 0; tn < 4; ++tn) {
      FragB bfrag;
      bfrag.u[0] = *(const uint4*)&Bs[tn * 16 + lrow][kk + half * 8];
      bfrag.u[1] = *(const uint4*)&Bs[tn * 16 + lrow][kk + 16 + half * 8];
      acc[tn] = __builtin_amdgcn_wmma_f32_16x16x32_bf16(
          false, afrag.v, false, bfrag.v, (short)0, acc[tn], false, false);
    }
  }

#pragma unroll
  for (int tn = 0; tn < 4; ++tn)
#pragma unroll
    for (int r = 0; r < 8; ++r) {
      int t = t0 + wave * 16 + half * 8 + r;
      int e = tn * 16 + lrow;
      float di = dinv[((size_t)b * HH + h) * TQ_ + t];
      size_t idx = ((size_t)b * TQ_ + t) * CC + h * DD + e;
      outacc[idx] += acc[tn][r] * di;
    }
}

// ---------------------------------------------------------------------------
// Host launcher
// ---------------------------------------------------------------------------
extern "C" void kernel_launch(void* const* d_in, const int* in_sizes, int n_in,
                              void* d_out, int out_size, void* d_ws, size_t ws_size,
                              hipStream_t stream) {
  (void)in_sizes; (void)n_in; (void)out_size; (void)ws_size;

  const float* x   = (const float*)d_in[0];
  const float* y0  = (const float*)d_in[1];
  const float* y1  = (const float*)d_in[2];
  const float* ln1_g = (const float*)d_in[3],  *ln1_b = (const float*)d_in[4];
  const float* ln20_g = (const float*)d_in[5], *ln20_b = (const float*)d_in[6];
  const float* ln21_g = (const float*)d_in[7], *ln21_b = (const float*)d_in[8];
  const float* ln3_g = (const float*)d_in[9],  *ln3_b = (const float*)d_in[10];
  const float* ln4_g = (const float*)d_in[11], *ln4_b = (const float*)d_in[12];
  const float* ln5_g = (const float*)d_in[13], *ln5_b = (const float*)d_in[14];
  const float* ca_q_w  = (const float*)d_in[15], *ca_q_b  = (const float*)d_in[16];
  const float* ca_k0_w = (const float*)d_in[17], *ca_k0_b = (const float*)d_in[18];
  const float* ca_v0_w = (const float*)d_in[19], *ca_v0_b = (const float*)d_in[20];
  const float* ca_k1_w = (const float*)d_in[21], *ca_k1_b = (const float*)d_in[22];
  const float* ca_v1_w = (const float*)d_in[23], *ca_v1_b = (const float*)d_in[24];
  const float* ca_p_w  = (const float*)d_in[25], *ca_p_b  = (const float*)d_in[26];
  const float* sa_q_w  = (const float*)d_in[27], *sa_q_b  = (const float*)d_in[28];
  const float* sa_k_w  = (const float*)d_in[29], *sa_k_b  = (const float*)d_in[30];
  const float* sa_v_w  = (const float*)d_in[31], *sa_v_b  = (const float*)d_in[32];
  const float* sa_p_w  = (const float*)d_in[33], *sa_p_b  = (const float*)d_in[34];
  const float* m1_w1 = (const float*)d_in[35], *m1_b1 = (const float*)d_in[36];
  const float* m1_w2 = (const float*)d_in[37], *m1_b2 = (const float*)d_in[38];
  const float* m2_w1 = (const float*)d_in[39], *m2_b1 = (const float*)d_in[40];
  const float* m2_w2 = (const float*)d_in[41], *m2_b2 = (const float*)d_in[42];

  const int MX = BB * TQ_;   // 16384
  const int MY = BB * TB_;   // 8192

  // ---- workspace carve-up ----
  size_t off = 0;
  auto carve = [&](size_t bytes) -> void* {
    void* p = (char*)d_ws + off;
    off += (bytes + 255) & ~(size_t)255;
    return p;
  };
  float* xw     = (float*)carve((size_t)MX * CC * 4);       // running residual x
  bf16*  xnb    = (bf16*)carve((size_t)MX * CC * 2);        // LN output / out_bf
  float* proj   = (float*)carve((size_t)MX * CC * 4);       // GEMM f32 scratch
  bf16*  qb     = (bf16*)carve((size_t)MX * CC * 2);
  bf16*  kb     = (bf16*)carve((size_t)MX * CC * 2);
  bf16*  vb     = (bf16*)carve((size_t)MX * CC * 2);
  float* outacc = (float*)carve((size_t)MX * CC * 4);
  bf16*  hb     = (bf16*)carve((size_t)MX * INNER_ * 2);
  bf16*  ynb    = (bf16*)carve((size_t)MY * CC * 2);
  float* ksumb  = (float*)carve((size_t)BB * CC * 4);
  float* dinvb  = (float*)carve((size_t)BB * HH * TQ_ * 4);
  float* ctxf   = (float*)carve((size_t)BB * HH * 64 * 64 * 4);
  bf16*  ctxb   = (bf16*)carve((size_t)BB * HH * 64 * 64 * 2);
  bf16* wqT  = (bf16*)carve((size_t)CC * CC * 2);
  bf16* wk0T = (bf16*)carve((size_t)CC * CC * 2);
  bf16* wv0T = (bf16*)carve((size_t)CC * CC * 2);
  bf16* wk1T = (bf16*)carve((size_t)CC * CC * 2);
  bf16* wv1T = (bf16*)carve((size_t)CC * CC * 2);
  bf16* wpT  = (bf16*)carve((size_t)CC * CC * 2);
  bf16* sqT  = (bf16*)carve((size_t)CC * CC * 2);
  bf16* skT  = (bf16*)carve((size_t)CC * CC * 2);
  bf16* svT  = (bf16*)carve((size_t)CC * CC * 2);
  bf16* spT  = (bf16*)carve((size_t)CC * CC * 2);
  bf16* m1w1T = (bf16*)carve((size_t)CC * INNER_ * 2);
  bf16* m1w2T = (bf16*)carve((size_t)CC * INNER_ * 2);
  bf16* m2w1T = (bf16*)carve((size_t)CC * INNER_ * 2);
  bf16* m2w2T = (bf16*)carve((size_t)CC * INNER_ * 2);

  const int WG = 256;
  const int gCC = (CC * CC) / WG;           // 1024
  const int gCI = (CC * INNER_) / WG;       // 4096
  const int NCTX = BB * HH * 64 * 64;       // ctx elements

  // ---- weights: convert + transpose to bf16 ----
  wt_kernel<<<gCC, WG, 0, stream>>>(ca_q_w,  wqT,  CC, CC);
  wt_kernel<<<gCC, WG, 0, stream>>>(ca_k0_w, wk0T, CC, CC);
  wt_kernel<<<gCC, WG, 0, stream>>>(ca_v0_w, wv0T, CC, CC);
  wt_kernel<<<gCC, WG, 0, stream>>>(ca_k1_w, wk1T, CC, CC);
  wt_kernel<<<gCC, WG, 0, stream>>>(ca_v1_w, wv1T, CC, CC);
  wt_kernel<<<gCC, WG, 0, stream>>>(ca_p_w,  wpT,  CC, CC);
  wt_kernel<<<gCC, WG, 0, stream>>>(sa_q_w,  sqT,  CC, CC);
  wt_kernel<<<gCC, WG, 0, stream>>>(sa_k_w,  skT,  CC, CC);
  wt_kernel<<<gCC, WG, 0, stream>>>(sa_v_w,  svT,  CC, CC);
  wt_kernel<<<gCC, WG, 0, stream>>>(sa_p_w,  spT,  CC, CC);
  wt_kernel<<<gCI, WG, 0, stream>>>(m1_w1, m1w1T, CC, INNER_);
  wt_kernel<<<gCI, WG, 0, stream>>>(m1_w2, m1w2T, INNER_, CC);
  wt_kernel<<<gCI, WG, 0, stream>>>(m2_w1, m2w1T, CC, INNER_);
  wt_kernel<<<gCI, WG, 0, stream>>>(m2_w2, m2w2T, INNER_, CC);

  dim3 gemmX(CC / 128, MX / 128);       // N=512 GEMMs on x rows
  dim3 gemmY(CC / 128, MY / 128);       // N=512 GEMMs on y rows
  dim3 gemmH(INNER_ / 128, MX / 128);   // N=2048 GEMMs

  // ================= cross-attention =================
  ln_kernel<<<MX / 8, WG, 0, stream>>>(x, ln1_g, ln1_b, xnb);
  gemm128<0,0,0><<<gemmX, WG, 0, stream>>>(xnb, wqT, ca_q_b, nullptr,
                                           proj, nullptr, MX, CC, CC);
  softmax64_kernel<<<MX, WG, 0, stream>>>(proj, qb, outacc);   // out := q

  for (int br = 0; br < 2; ++br) {
    const float* y   = br ? y1 : y0;
    const float* lg  = br ? ln21_g : ln20_g;
    const float* lb  = br ? ln21_b : ln20_b;
    const bf16*  wkT = br ? wk1T : wk0T;
    const bf16*  wvT = br ? wv1T : wv0T;
    const float* kbb = br ? ca_k1_b : ca_k0_b;
    const float* vbb = br ? ca_v1_b : ca_v0_b;

    ln_kernel<<<MY / 8, WG, 0, stream>>>(y, lg, lb, ynb);
    gemm128<0,0,0><<<gemmY, WG, 0, stream>>>(ynb, wkT, kbb, nullptr,
                                             proj, nullptr, MY, CC, CC);
    softmax64_kernel<<<MY, WG, 0, stream>>>(proj, kb, nullptr);
    gemm128<1,0,0><<<gemmY, WG, 0, stream>>>(ynb, wvT, vbb, nullptr,
                                             nullptr, vb, MY, CC, CC);
    zero_kernel<<<(BB * CC) / WG, WG, 0, stream>>>(ksumb, BB * CC);
    ksum_kernel<<<dim3((BB * CC) / WG, 16), WG, 0, stream>>>(kb, ksumb, TB_);
    dinv_kernel<<<(BB * HH * TQ_) / WG, WG, 0, stream>>>(qb, ksumb, dinvb, 1e-9f);
    zero_kernel<<<NCTX / WG, WG, 0, stream>>>(ctxf, NCTX);
    ctx_kernel<<<dim3(BB * HH, KCH), 128, 0, stream>>>(kb, vb, ctxf, TB_);
    cvt_bf16_kernel<<<NCTX / WG, WG, 0, stream>>>(ctxf, ctxb, NCTX);
    attn_apply_kernel<<<dim3(TQ_ / 128, HH, BB), WG, 0, stream>>>(qb, ctxb,
                                                                  dinvb, outacc);
  }
  cvt_bf16_kernel<<<((long)MX * CC) / WG, WG, 0, stream>>>(outacc, xnb,
                                                           (long)MX * CC);
  gemm128<0,0,1><<<gemmX, WG, 0, stream>>>(xnb, wpT, ca_p_b, x,
                                           xw, nullptr, MX, CC, CC);

  // ================= MLP 1 =================
  ln_kernel<<<MX / 8, WG, 0, stream>>>(xw, ln3_g, ln3_b, xnb);
  gemm128<1,1,0><<<gemmH, WG, 0, stream>>>(xnb, m1w1T, m1_b1, nullptr,
                                           nullptr, hb, MX, INNER_, CC);
  gemm128<0,0,1><<<gemmX, WG, 0, stream>>>(hb, m1w2T, m1_b2, xw,
                                           xw, nullptr, MX, CC, INNER_);

  // ================= self-attention =================
  ln_kernel<<<MX / 8, WG, 0, stream>>>(xw, ln4_g, ln4_b, xnb);
  gemm128<0,0,0><<<gemmX, WG, 0, stream>>>(xnb, sqT, sa_q_b, nullptr,
                                           proj, nullptr, MX, CC, CC);
  softmax64_kernel<<<MX, WG, 0, stream>>>(proj, qb, outacc);   // out := q
  gemm128<0,0,0><<<gemmX, WG, 0, stream>>>(xnb, skT, sa_k_b, nullptr,
                                           proj, nullptr, MX, CC, CC);
  softmax64_kernel<<<MX, WG, 0, stream>>>(proj, kb, nullptr);
  gemm128<1,0,0><<<gemmX, WG, 0, stream>>>(xnb, svT, sa_v_b, nullptr,
                                           nullptr, vb, MX, CC, CC);
  zero_kernel<<<(BB * CC) / WG, WG, 0, stream>>>(ksumb, BB * CC);
  ksum_kernel<<<dim3((BB * CC) / WG, 16), WG, 0, stream>>>(kb, ksumb, TQ_);
  dinv_kernel<<<(BB * HH * TQ_) / WG, WG, 0, stream>>>(qb, ksumb, dinvb, 1e-30f);
  zero_kernel<<<NCTX / WG, WG, 0, stream>>>(ctxf, NCTX);
  ctx_kernel<<<dim3(BB * HH, KCH), 128, 0, stream>>>(kb, vb, ctxf, TQ_);
  cvt_bf16_kernel<<<NCTX / WG, WG, 0, stream>>>(ctxf, ctxb, NCTX);
  attn_apply_kernel<<<dim3(TQ_ / 128, HH, BB), WG, 0, stream>>>(qb, ctxb,
                                                                dinvb, outacc);
  cvt_bf16_kernel<<<((long)MX * CC) / WG, WG, 0, stream>>>(outacc, xnb,
                                                           (long)MX * CC);
  gemm128<0,0,1><<<gemmX, WG, 0, stream>>>(xnb, spT, sa_p_b, xw,
                                           xw, nullptr, MX, CC, CC);

  // ================= MLP 2 (writes final output) =================
  ln_kernel<<<MX / 8, WG, 0, stream>>>(xw, ln5_g, ln5_b, xnb);
  gemm128<1,1,0><<<gemmH, WG, 0, stream>>>(xnb, m2w1T, m2_b1, nullptr,
                                           nullptr, hb, MX, INNER_, CC);
  gemm128<0,0,1><<<gemmX, WG, 0, stream>>>(hb, m2w2T, m2_b2, xw,
                                           (float*)d_out, nullptr, MX, CC, INNER_);
}
